// Encoder_4466765987988
// MI455X (gfx1250) — matrix operand order repaired and checked by reading, not compiled, not run on hardware
//
#include <hip/hip_runtime.h>

// ---------------------------------------------------------------------------
// Transformer encoder forward for MI455X (gfx1250, wave32, WMMA).
// All GEMMs run as bf16 WMMA (v_wmma_f32_16x16x32_bf16) with fp32 accumulate.
// GEMM uses double-buffered LDS tiles: A staged via global_load_async_to_lds
// (ASYNCcnt) issued ahead of compute, B staged transposed so fragments are
// ds_load_b128.  Softmax / LayerNorm / residuals stay in fp32.
// ---------------------------------------------------------------------------

typedef __attribute__((ext_vector_type(16))) __bf16 v16bf;
typedef __attribute__((ext_vector_type(8)))  float  v8f;

#define NLAYER 6
#define NH     8
#define DMODEL 512
#define DHEAD  64
#define FFDIM  1024
#define SEQLEN 256
#define NBATCH 16
#define NTOK   (NBATCH * SEQLEN)   // 4096
#define QKVN   (3 * DMODEL)        // 1536

static __device__ __forceinline__ __bf16 f2bf(float f) {
    unsigned int u = __builtin_bit_cast(unsigned int, f);
    u += 0x7FFFu + ((u >> 16) & 1u);                     // round-to-nearest-even
    return __builtin_bit_cast(__bf16, (unsigned short)(u >> 16));
}

union Frag16 { uint4 q[2]; v16bf v; };                   // 16 x bf16 fragment
union Pack8  { uint4 q;    __bf16 h[8]; };               // 8 x bf16 payload

// low 32 bits of a generic LDS pointer == LDS byte offset (aperture layout)
static __device__ __forceinline__ unsigned lds_addr32(const void* p) {
    return (unsigned)(unsigned long long)p;
}

// Async copy of 16 bytes global -> LDS, tracked by ASYNCcnt.
static __device__ __forceinline__ void async_g2l_b128(unsigned lds_dst,
                                                      unsigned voff,
                                                      unsigned long long gbase) {
    asm volatile("global_load_async_to_lds_b128 %0, %1, %2"
                 :: "v"(lds_dst), "v"(voff), "s"(gbase) : "memory");
}
static __device__ __forceinline__ void wait_asynccnt0() {
    asm volatile("s_wait_asynccnt 0x0" ::: "memory");
}

// ---------------------------------------------------------------------------
// Weight repack: fp32 -> bf16, layouts for [M,K] x [K,N].
// ---------------------------------------------------------------------------

// Interleave w_qs/w_ks/w_vs [NL,H,D,DK] into per-layer [D, 3*D], column
// n = sel*512 + h*64 + k.  1/sqrt(D) temperature folded into W_q.
__global__ __launch_bounds__(256)
void repack_qkv_kernel(const float* __restrict__ wq, const float* __restrict__ wk,
                       const float* __restrict__ wv, __bf16* __restrict__ out) {
    int idx = blockIdx.x * 256 + threadIdx.x;
    const int per_sel = NH * DMODEL * DHEAD;          // 262144
    const int per_l   = 3 * per_sel;                  // 786432
    if (idx >= NLAYER * per_l) return;
    int l   = idx / per_l;
    int rem = idx - l * per_l;
    int sel = rem / per_sel;
    int r2  = rem - sel * per_sel;
    int h   = r2 >> 15;
    int d   = (r2 >> 6) & 511;
    int kk  = r2 & 63;
    const float* src = (sel == 0) ? wq : (sel == 1) ? wk : wv;
    float v = src[(((size_t)l * NH + h) * DMODEL + d) * DHEAD + kk];
    if (sel == 0) v *= 0.04419417382415922f;          // 1/sqrt(512)
    out[(size_t)l * DMODEL * QKVN + (size_t)d * QKVN + sel * DMODEL + h * DHEAD + kk] = f2bf(v);
}

// out[c*R + r] = (bf16) in[r*C + c]
__global__ __launch_bounds__(256)
void transpose_bf16_kernel(const float* __restrict__ in, __bf16* __restrict__ out,
                           int R, int C) {
    int idx = blockIdx.x * 256 + threadIdx.x;
    if (idx >= R * C) return;
    int r = idx / C, c = idx - r * C;
    out[(size_t)c * R + r] = f2bf(in[idx]);
}

// ---------------------------------------------------------------------------
// Embedding: x = word_emb[seq] + pos_emb[pos]; emit fp32 + bf16.
// ---------------------------------------------------------------------------
__global__ __launch_bounds__(256)
void embed_kernel(const int* __restrict__ seq, const int* __restrict__ pos,
                  const float* __restrict__ wemb, const float* __restrict__ pemb,
                  float* __restrict__ xf, __bf16* __restrict__ xb) {
    int token = blockIdx.x;
    int tid   = threadIdx.x;
    int s = seq[token], p = pos[token];
    size_t base = (size_t)token * DMODEL;
    float v0 = wemb[(size_t)s * DMODEL + tid]       + pemb[(size_t)p * DMODEL + tid];
    float v1 = wemb[(size_t)s * DMODEL + 256 + tid] + pemb[(size_t)p * DMODEL + 256 + tid];
    xf[base + tid]       = v0;  xb[base + tid]       = f2bf(v0);
    xf[base + 256 + tid] = v1;  xb[base + 256 + tid] = f2bf(v1);
}

// ---------------------------------------------------------------------------
// bf16 WMMA GEMM: C[M,N] = A[M,K] * B[K,N] (+bias) (ReLU).
// Block tile 128x64, 8 waves, each wave owns a 16x64 strip (4 WMMA acc).
// Double-buffered LDS: next A tile DMA'd (async->LDS) and next B tile loaded
// to registers BEFORE computing the current tile; wait/barrier after compute.
// ---------------------------------------------------------------------------
template <bool HAS_BIAS, bool RELU, bool WF, bool WB>
__global__ __launch_bounds__(256)
void gemm_bf16_kernel(const __bf16* __restrict__ A, const __bf16* __restrict__ B,
                      const float* __restrict__ bias,
                      float* __restrict__ Cf, __bf16* __restrict__ Cb,
                      int N, int K) {
    __shared__ __align__(16) __bf16 As[2][128][32];   // 16 KB ping-pong
    __shared__ __align__(16) __bf16 Bt[2][64][32];    //  8 KB ping-pong, Bt[n][k]

    const int bm   = blockIdx.x * 128;
    const int bn   = blockIdx.y * 64;
    const int tid  = threadIdx.x;
    const int wave = tid >> 5;         // 16-row strip owner
    const int lane = tid & 31;

    const int lr     = lane & 15;
    const int kbaseA = (lane >> 4) * 8;    // A fragment K split (ISA 7.12.2)
    const int kbaseB = (lane >> 4) * 8;    // B fragment K split

    // staging coords: A = 128x32 (two b128 chunks / thread), B = 32x64
    const int br_s = tid >> 3, bc_s = (tid & 7) * 8;
    const unsigned long long Abase = (unsigned long long)A;

    v8f acc[4] = {};

    // ---- prologue: stage buffer 0 ----
#pragma unroll
    for (int i = 0; i < 2; ++i) {
        int cid = tid * 2 + i;                  // 0..511
        int r = cid >> 2, c = (cid & 3) * 8;
        async_g2l_b128(lds_addr32(&As[0][r][c]),
                       (unsigned)(((unsigned)(bm + r) * (unsigned)K + (unsigned)c) * 2u),
                       Abase);
    }
    {
        Pack8 u;
        u.q = *(const uint4*)(B + (size_t)br_s * N + bn + bc_s);
#pragma unroll
        for (int j = 0; j < 8; ++j) Bt[0][bc_s + j][br_s] = u.h[j];
    }
    wait_asynccnt0();
    __syncthreads();

    const int nsteps = K >> 5;
    for (int s = 0; s < nsteps; ++s) {
        const int cur = s & 1, nxt = cur ^ 1;
        const bool have_next = (s + 1) < nsteps;
        const int k0n = (s + 1) << 5;

        Pack8 u = {};
        if (have_next) {
            // prefetch next tiles: A via async DMA, B into registers
#pragma unroll
            for (int i = 0; i < 2; ++i) {
                int cid = tid * 2 + i;
                int r = cid >> 2, c = (cid & 3) * 8;
                async_g2l_b128(lds_addr32(&As[nxt][r][c]),
                               (unsigned)(((unsigned)(bm + r) * (unsigned)K + (unsigned)(k0n + c)) * 2u),
                               Abase);
            }
            u.q = *(const uint4*)(B + (size_t)(k0n + br_s) * N + bn + bc_s);
        }

        // ---- compute on current buffer (overlaps in-flight DMA) ----
        Frag16 fa;
        fa.q[0] = *(const uint4*)(&As[cur][wave * 16 + lr][kbaseA]);
        fa.q[1] = *(const uint4*)(&As[cur][wave * 16 + lr][16 + kbaseA]);
#pragma unroll
        for (int t = 0; t < 4; ++t) {
            Frag16 fb;
            fb.q[0] = *(const uint4*)(&Bt[cur][t * 16 + lr][kbaseB]);
            fb.q[1] = *(const uint4*)(&Bt[cur][t * 16 + lr][kbaseB + 16]);
            acc[t] = __builtin_amdgcn_wmma_f32_16x16x32_bf16(false, fa.v, false, fb.v,
                                                             (short)0, acc[t], false, false);
        }

        if (have_next) {
#pragma unroll
            for (int j = 0; j < 8; ++j) Bt[nxt][bc_s + j][br_s] = u.h[j];
            wait_asynccnt0();
        }
        __syncthreads();
    }

    // C layout: VGPR j -> row j (lanes 0-15) / j+8 (lanes 16-31), col = lane&15
    const int lm = (lane >> 4) * 8;
    const int lc = lane & 15;
#pragma unroll
    for (int t = 0; t < 4; ++t) {
        const int ct = bn + t * 16 + lc;
        const float bt = HAS_BIAS ? bias[ct] : 0.0f;
#pragma unroll
        for (int j = 0; j < 8; ++j) {
            int row = bm + wave * 16 + lm + j;
            float v = acc[t][j] + bt;
            if (RELU) v = fmaxf(v, 0.0f);
            size_t rb = (size_t)row * N;
            if (WF) Cf[rb + ct] = v;
            if (WB) Cb[rb + ct] = f2bf(v);
        }
    }
}

// ---------------------------------------------------------------------------
// Attention: one block = (32 query rows, head h, batch b).
// Phase 1: S = Q K^T (WMMA, scale pre-folded), pad-mask, fp32 softmax.
// Phase 2: O = P V with V slab staged transposed in LDS per K-step.
// ---------------------------------------------------------------------------
__global__ __launch_bounds__(256)
void attn_kernel(const __bf16* __restrict__ qkv, const int* __restrict__ seq,
                 float* __restrict__ attn_out, __bf16* __restrict__ oconcat) {
    __shared__ float  S[32][SEQLEN];                 // 32 KB
    __shared__ __bf16 Pb[32][SEQLEN];                // 16 KB
    __shared__ __align__(16) __bf16 Vt[64][32];      //  4 KB (V^T slab)
    __shared__ float  red[32][8];
    __shared__ float  rowmax[32], rowinv[32];

    const int qt = blockIdx.x;      // 0..7
    const int h  = blockIdx.y;      // 0..7
    const int b  = blockIdx.z;      // 0..15
    const int tid = threadIdx.x, wave = tid >> 5, lane = tid & 31;
    const int q0 = qt * 32;

    const __bf16* Qp = qkv + (size_t)(b * SEQLEN) * QKVN + h * DHEAD;
    const __bf16* Kp = Qp + DMODEL;
    const __bf16* Vp = Qp + 2 * DMODEL;

    const int lr     = lane & 15;
    const int kbaseA = (lane >> 4) * 8;
    const int kbaseB = (lane >> 4) * 8;
    const int lm     = (lane >> 4) * 8;

    // ---- Phase 1: logits (Q 32x64 . K^T 64x256) ----
    {
        const int tr    = wave >> 2;            // query sub-tile
        const int cbase = (wave & 3) * 64;      // key column strip
        v8f acc[4] = {};
        for (int k0 = 0; k0 < DHEAD; k0 += 32) {
            Frag16 aq;
            const __bf16* qrow = Qp + (size_t)(q0 + tr * 16 + lr) * QKVN + k0;
            aq.q[0] = *(const uint4*)(qrow + kbaseA);
            aq.q[1] = *(const uint4*)(qrow + 16 + kbaseA);
#pragma unroll
            for (int t = 0; t < 4; ++t) {
                const __bf16* krow = Kp + (size_t)(cbase + t * 16 + lr) * QKVN + k0;
                Frag16 bk;                       // B = K^T: two 16B chunks per lane (k 8h.., 16+8h..)
                bk.q[0] = *(const uint4*)(krow + kbaseB);
                bk.q[1] = *(const uint4*)(krow + kbaseB + 16);
                acc[t] = __builtin_amdgcn_wmma_f32_16x16x32_bf16(false, aq.v, false, bk.v,
                                                                 (short)0, acc[t], false, false);
            }
        }
#pragma unroll
        for (int t = 0; t < 4; ++t) {
            int col = cbase + t * 16 + lr;
            bool masked = (seq[b * SEQLEN + col] == 0);
#pragma unroll
            for (int j = 0; j < 8; ++j)
                S[tr * 16 + lm + j][col] = masked ? -1e30f : acc[t][j];
        }
    }
    __syncthreads();

    // ---- softmax (8 segments of 32 keys per row) ----
    const int row = tid >> 3, seg = tid & 7;
    float mx = -3.4e38f;
    for (int c = seg * 32; c < seg * 32 + 32; ++c) mx = fmaxf(mx, S[row][c]);
    red[row][seg] = mx;
    __syncthreads();
    if (tid < 32) {
        float m = red[tid][0];
        for (int s = 1; s < 8; ++s) m = fmaxf(m, red[tid][s]);
        rowmax[tid] = m;
    }
    __syncthreads();
    float m = rowmax[row], sum = 0.0f;
    for (int c = seg * 32; c < seg * 32 + 32; ++c) {
        float e = __expf(S[row][c] - m);
        S[row][c] = e; sum += e;
    }
    red[row][seg] = sum;
    __syncthreads();
    if (tid < 32) {
        float s = 0.0f;
        for (int ss = 0; ss < 8; ++ss) s += red[tid][ss];
        rowinv[tid] = 1.0f / s;
    }
    __syncthreads();
    float inv = rowinv[row];
    float* arow = attn_out + ((size_t)(h * NBATCH + b) * SEQLEN + (q0 + row)) * SEQLEN;
    for (int c = seg * 32; c < seg * 32 + 32; ++c) {
        float p = S[row][c] * inv;
        __builtin_nontemporal_store(p, &arow[c]);   // streamed, never re-read
        Pb[row][c] = f2bf(p);
    }

    // ---- Phase 2: O = P(32x256) x V(256x64) ----
    {
        const int tr = wave & 1, tc = wave >> 1;     // 2x4 tiles, one per wave
        const int vr_s = tid >> 3, vc_s = (tid & 7) * 8;   // V staging coords
        v8f acc = {};
        for (int k0 = 0; k0 < SEQLEN; k0 += 32) {
            __syncthreads();
            {   // stage V slab transposed: Vt[dhead][key]
                Pack8 u;
                u.q = *(const uint4*)(Vp + (size_t)(k0 + vr_s) * QKVN + vc_s);
#pragma unroll
                for (int j = 0; j < 8; ++j) Vt[vc_s + j][vr_s] = u.h[j];
            }
            __syncthreads();
            Frag16 ap, bv;
            ap.q[0] = *(const uint4*)(&Pb[tr * 16 + lr][k0 + kbaseA]);
            ap.q[1] = *(const uint4*)(&Pb[tr * 16 + lr][k0 + 16 + kbaseA]);
            bv.q[0] = *(const uint4*)(&Vt[tc * 16 + lr][kbaseB]);
            bv.q[1] = *(const uint4*)(&Vt[tc * 16 + lr][kbaseB + 16]);
            acc = __builtin_amdgcn_wmma_f32_16x16x32_bf16(false, ap.v, false, bv.v,
                                                          (short)0, acc, false, false);
        }
        const int lc = lane & 15;
#pragma unroll
        for (int j = 0; j < 8; ++j) {
            int r = q0 + tr * 16 + lm + j;
            int c = h * DHEAD + tc * 16 + lc;        // head-major concat
            oconcat[(size_t)(b * SEQLEN + r) * DMODEL + c] = f2bf(acc[j]);
        }
    }
}

// ---------------------------------------------------------------------------
// Residual add + LayerNorm (mean + unbiased std, eps added to std).
// ---------------------------------------------------------------------------
__global__ __launch_bounds__(256)
void residual_ln_kernel(const float* __restrict__ y, float* __restrict__ x_f32,
                        const float* __restrict__ gain, const float* __restrict__ beta,
                        __bf16* __restrict__ x_bf16, float* __restrict__ extra) {
    __shared__ float red[256];
    const int token = blockIdx.x;
    const int tid   = threadIdx.x;
    const size_t base = (size_t)token * DMODEL;

    float z0 = y[base + tid]       + x_f32[base + tid];
    float z1 = y[base + 256 + tid] + x_f32[base + 256 + tid];

    red[tid] = z0 + z1;
    __syncthreads();
    for (int s = 128; s > 0; s >>= 1) {
        if (tid < s) red[tid] += red[tid + s];
        __syncthreads();
    }
    float mu = red[0] * (1.0f / DMODEL);
    __syncthreads();

    float d0 = z0 - mu, d1 = z1 - mu;
    red[tid] = d0 * d0 + d1 * d1;
    __syncthreads();
    for (int s = 128; s > 0; s >>= 1) {
        if (tid < s) red[tid] += red[tid + s];
        __syncthreads();
    }
    float sd  = sqrtf(red[0] * (1.0f / (DMODEL - 1)));   // ddof=1
    float inv = 1.0f / (sd + 1e-3f);

    float o0 = d0 * inv * gain[tid]       + beta[tid];
    float o1 = d1 * inv * gain[256 + tid] + beta[256 + tid];

    x_f32[base + tid]       = o0;  x_bf16[base + tid]       = f2bf(o0);
    x_f32[base + 256 + tid] = o1;  x_bf16[base + 256 + tid] = f2bf(o1);
    if (extra) {
        __builtin_nontemporal_store(o0, &extra[base + tid]);
        __builtin_nontemporal_store(o1, &extra[base + 256 + tid]);
    }
}

// ---------------------------------------------------------------------------
// Host orchestration.
// ---------------------------------------------------------------------------
extern "C" void kernel_launch(void* const* d_in, const int* in_sizes, int n_in,
                              void* d_out, int out_size, void* d_ws, size_t ws_size,
                              hipStream_t stream) {
    (void)in_sizes; (void)n_in; (void)out_size; (void)ws_size;

    const int*   src_seq  = (const int*)  d_in[0];
    const int*   src_pos  = (const int*)  d_in[1];
    const float* word_emb = (const float*)d_in[2];
    const float* pos_emb  = (const float*)d_in[3];
    const float* w_qs     = (const float*)d_in[4];
    const float* w_ks     = (const float*)d_in[5];
    const float* w_vs     = (const float*)d_in[6];
    const float* proj_w   = (const float*)d_in[7];
    const float* proj_b   = (const float*)d_in[8];
    const float* ln1_a    = (const float*)d_in[9];
    const float* ln1_b    = (const float*)d_in[10];
    const float* ffn_w1   = (const float*)d_in[11];
    const float* ffn_b1   = (const float*)d_in[12];
    const float* ffn_w2   = (const float*)d_in[13];
    const float* ffn_b2   = (const float*)d_in[14];
    const float* ln2_a    = (const float*)d_in[15];
    const float* ln2_b    = (const float*)d_in[16];

    char* ws = (char*)d_ws;
    size_t off = 0;
    auto carve = [&](size_t bytes) { char* p = ws + off; off += (bytes + 255) & ~(size_t)255; return p; };
    float*  x_f32    = (float*) carve((size_t)NTOK * DMODEL * 4);
    __bf16* x_bf16   = (__bf16*)carve((size_t)NTOK * DMODEL * 2);
    __bf16* qkv      = (__bf16*)carve((size_t)NTOK * QKVN   * 2);
    __bf16* oconcat  = (__bf16*)carve((size_t)NTOK * DMODEL * 2);
    __bf16* h_bf16   = (__bf16*)carve((size_t)NTOK * FFDIM  * 2);
    float*  gemm_f32 = (float*) carve((size_t)NTOK * DMODEL * 4);
    __bf16* qkvw     = (__bf16*)carve((size_t)NLAYER * DMODEL * QKVN  * 2);
    __bf16* projt    = (__bf16*)carve((size_t)NLAYER * DMODEL * DMODEL * 2);
    __bf16* w1t      = (__bf16*)carve((size_t)NLAYER * DMODEL * FFDIM * 2);
    __bf16* w2t      = (__bf16*)carve((size_t)NLAYER * FFDIM  * DMODEL * 2);

    dim3 blk(256);

    {   // weight repacks (deterministic, every call)
        int total = NLAYER * 3 * NH * DMODEL * DHEAD;
        repack_qkv_kernel<<<(total + 255) / 256, blk, 0, stream>>>(w_qs, w_ks, w_vs, qkvw);
        for (int l = 0; l < NLAYER; ++l) {
            transpose_bf16_kernel<<<(DMODEL * DMODEL + 255) / 256, blk, 0, stream>>>(
                proj_w + (size_t)l * DMODEL * DMODEL, projt + (size_t)l * DMODEL * DMODEL,
                DMODEL, DMODEL);
            transpose_bf16_kernel<<<(FFDIM * DMODEL + 255) / 256, blk, 0, stream>>>(
                ffn_w1 + (size_t)l * FFDIM * DMODEL, w1t + (size_t)l * DMODEL * FFDIM,
                FFDIM, DMODEL);
            transpose_bf16_kernel<<<(DMODEL * FFDIM + 255) / 256, blk, 0, stream>>>(
                ffn_w2 + (size_t)l * DMODEL * FFDIM, w2t + (size_t)l * FFDIM * DMODEL,
                DMODEL, FFDIM);
        }
    }

    embed_kernel<<<NTOK, blk, 0, stream>>>(src_seq, src_pos, word_emb, pos_emb, x_f32, x_bf16);

    float* outs  = (float*)d_out;                                      // [NL,B,L,D]
    float* attns = (float*)d_out + (size_t)NLAYER * NTOK * DMODEL;     // [NL,H*B,L,L]

    for (int l = 0; l < NLAYER; ++l) {
        // fused QKV projection: [4096,512] x [512,1536] -> bf16
        gemm_bf16_kernel<false, false, false, true>
            <<<dim3(NTOK / 128, QKVN / 64), blk, 0, stream>>>(
            x_bf16, qkvw + (size_t)l * DMODEL * QKVN, nullptr,
            nullptr, qkv, QKVN, DMODEL);

        attn_kernel<<<dim3(SEQLEN / 32, NH, NBATCH), blk, 0, stream>>>(
            qkv, src_seq,
            attns + (size_t)l * NH * NBATCH * SEQLEN * SEQLEN, oconcat);

        // output projection: [4096,512] x [512,512] + b -> fp32
        gemm_bf16_kernel<true, false, true, false>
            <<<dim3(NTOK / 128, DMODEL / 64), blk, 0, stream>>>(
            oconcat, projt + (size_t)l * DMODEL * DMODEL, proj_b + (size_t)l * DMODEL,
            gemm_f32, nullptr, DMODEL, DMODEL);

        residual_ln_kernel<<<NTOK, blk, 0, stream>>>(
            gemm_f32, x_f32, ln1_a + (size_t)l * DMODEL, ln1_b + (size_t)l * DMODEL,
            x_bf16, nullptr);

        // FFN1 + ReLU: [4096,512] x [512,1024] -> bf16
        gemm_bf16_kernel<true, true, false, true>
            <<<dim3(NTOK / 128, FFDIM / 64), blk, 0, stream>>>(
            x_bf16, w1t + (size_t)l * DMODEL * FFDIM, ffn_b1 + (size_t)l * FFDIM,
            nullptr, h_bf16, FFDIM, DMODEL);

        // FFN2: [4096,1024] x [1024,512] -> fp32
        gemm_bf16_kernel<true, false, true, false>
            <<<dim3(NTOK / 128, DMODEL / 64), blk, 0, stream>>>(
            h_bf16, w2t + (size_t)l * FFDIM * DMODEL, ffn_b2 + (size_t)l * DMODEL,
            gemm_f32, nullptr, DMODEL, FFDIM);

        residual_ln_kernel<<<NTOK, blk, 0, stream>>>(
            gemm_f32, x_f32, ln2_a + (size_t)l * DMODEL, ln2_b + (size_t)l * DMODEL,
            x_bf16, outs + (size_t)l * NTOK * DMODEL);
    }
}
